// NTXentLoss_249108103545
// MI455X (gfx1250) — matrix operand order repaired and checked by reading, not compiled, not run on hardware
//
#include <hip/hip_runtime.h>

typedef __attribute__((ext_vector_type(16))) __bf16 v16bf;
typedef __attribute__((ext_vector_type(8)))  __bf16 v8bf;
typedef __attribute__((ext_vector_type(8)))  float  v8f;
typedef __attribute__((ext_vector_type(4)))  unsigned int v4u;
typedef __attribute__((ext_vector_type(4)))  int    v4i;
typedef __attribute__((ext_vector_type(8)))  int    v8i;

#if __has_builtin(__builtin_amdgcn_exp2f)
#define EXP2F(x) __builtin_amdgcn_exp2f(x)
#else
#define EXP2F(x) exp2f(x)
#endif
#if __has_builtin(__builtin_amdgcn_logf)
#define LOG2F(x) __builtin_amdgcn_logf(x)
#else
#define LOG2F(x) log2f(x)
#endif

#define NROWS   8192                 // 2b
#define HALF_B  4096                 // b
#define DIM     256                  // d
#define NTILES  (NROWS / 16)         // 512 column tiles
#define NEGBIG  (-1.0e30f)
#define PRESCALE 1.6986436f          // sqrt((1/T)*log2 e), T=0.5
#define LN2      0.6931471805599453f

// compare-select max: avoids v_max_num canonicalization pairs, dual-issuable
__device__ __forceinline__ float smax(float a, float b) { return a > b ? a : b; }

// ---------------------------------------------------------------------------
// Kernel 1: reps = concat(zjs, zis), pre-scaled so the bf16 GEMM directly
// produces logits in the log2 domain; f32 -> bf16 RNE.
// ---------------------------------------------------------------------------
__global__ void ntxent_convert(const float* __restrict__ zis,
                               const float* __restrict__ zjs,
                               unsigned short* __restrict__ repsB) {
    int idx = blockIdx.x * blockDim.x + threadIdx.x;
    if (idx >= NROWS * DIM) return;
    float x = (idx < HALF_B * DIM) ? zjs[idx] : zis[idx - HALF_B * DIM];
    x *= PRESCALE;
    unsigned int u = __float_as_uint(x);
    unsigned int r = (u + 0x7FFFu + ((u >> 16) & 1u)) >> 16;   // RNE
    repsB[idx] = (unsigned short)r;
}

// ---------------------------------------------------------------------------
// TDM: one tensor_load_to_lds moves a whole 16x256 bf16 tile (8 KB).
// Descriptor per CDNA5 ISA (2-D, data_size=2B, type=2).
// ---------------------------------------------------------------------------
#if __has_builtin(__builtin_amdgcn_tensor_load_to_lds)
#define HAVE_TDM 1
__device__ __forceinline__ void tdm_tile_load(unsigned ldsOff,
                                              const unsigned short* gsrc) {
    unsigned long long ga = (unsigned long long)(uintptr_t)gsrc;
    v4u g0 = { 1u,                                        // count=1 (valid D#)
               ldsOff,                                    // lds_addr
               (unsigned)(ga & 0xFFFFFFFFu),              // global_addr[31:0]
               (unsigned)((ga >> 32) & 0x1FFFFFFu) | (2u << 30) }; // [56:32]|type=2
    v8i g1 = { (int)0x00010000,        // workgroup_mask=0, data_size=1 (2 bytes)
               (int)(DIM << 16),       // tensor_dim0 = 256 (lo16 in [31:16])
               (int)(NROWS << 16),     // tensor_dim0 hi=0 | tensor_dim1 lo16 = 8192
               (int)(DIM << 16),       // tensor_dim1 hi=0 | tile_dim0 = 256
               16,                     // tile_dim1 = 16, tile_dim2 = 0
               DIM,                    // tensor_dim0_stride = 256 (lo32)
               (int)(DIM << 16),       // stride0 hi=0 | tensor_dim1_stride lo16 = 256
               0 };                    // tensor_dim1_stride hi
    v4i gz = { 0, 0, 0, 0 };
#if __has_include(<hip/amd_detail/amd_gfx1250_TDM.h>)
    v8i gz8 = { 0, 0, 0, 0, 0, 0, 0, 0 };
    __builtin_amdgcn_tensor_load_to_lds(g0, g1, gz, gz, gz8, 0);   // clang-23 form
#else
    __builtin_amdgcn_tensor_load_to_lds(g0, g1, gz, gz, 0);        // ROCm 7.2 form
#endif
}
__device__ __forceinline__ void tdm_wait0() {
#if __has_builtin(__builtin_amdgcn_s_wait_tensorcnt)
    __builtin_amdgcn_s_wait_tensorcnt(0);
#else
    asm volatile("s_wait_tensorcnt 0x0" ::: "memory");
#endif
}
#else
#define HAVE_TDM 0
#endif

// ---------------------------------------------------------------------------
// Kernel 2: fused sim = reps*reps^T (bf16 WMMA, f32 acc) + streaming
// logsumexp. One wave owns a 16-row strip (A resident in 64 VGPRs); 8 waves
// share TDM-double-buffered B tiles. The diagonal/positive tiles are handled
// on a SCALAR-branched general path (2 of 512 iterations, SGPR-uniform via
// readfirstlane); the hot path is comparison-free. accuracy = (pos == rowmax).
// ---------------------------------------------------------------------------
__global__ void __launch_bounds__(256)
ntxent_main(const unsigned short* __restrict__ repsB,
            float* __restrict__ partials) {
    __shared__ unsigned short tiles[2][16 * DIM];   // 2 x 8 KB double buffer
    __shared__ float red[8][2];

    const int t    = threadIdx.x;
    const int lane = t & 31;
    const int lr   = lane & 15;        // A row / B,C column within tile
    const int hi   = lane >> 4;        // K-half selector
    // wave id as an SGPR so R/diagT/posT are provably uniform -> s_cbranch
    const int wave = __builtin_amdgcn_readfirstlane(t >> 5);
    const int R    = (blockIdx.x * 8 + wave) * 16;
    const int diagT = R >> 4;                                 // diagonal tile
    const int posT  = (diagT + (NROWS >> 5)) & (NTILES - 1);  // +256 mod 512

    // ---- resident A strip: 8 K-steps of 16x32 bf16 fragments ----
    v16bf afrag[8];
    const unsigned short* arow = repsB + (size_t)(R + lr) * DIM;
    #pragma unroll
    for (int s = 0; s < 8; ++s) {
        const int kb = 32 * s;
        v8bf c0 = *(const v8bf*)(arow + kb + 8 * hi);
        v8bf c1 = *(const v8bf*)(arow + kb + 16 + 8 * hi);
        ((v8bf*)&afrag[s])[0] = c0;
        ((v8bf*)&afrag[s])[1] = c1;
    }

    // per-lane state for 8 row slots (row = R + v + 8*hi, column class lr)
    float m[8], ssum[8], pos[8];
    #pragma unroll
    for (int v = 0; v < 8; ++v) { m[v] = NEGBIG; ssum[v] = 0.0f; pos[v] = NEGBIG; }

#if HAVE_TDM
    if (wave == 0) tdm_tile_load((unsigned)(uintptr_t)&tiles[0][0], repsB);
#endif

    for (int ct = 0; ct < NTILES; ++ct) {
        const int cur = ct & 1;
        const int Cb  = ct * 16;
#if HAVE_TDM
        if (wave == 0) tdm_wait0();            // tile ct landed in LDS
        __syncthreads();                       // publish; all done reading buf^1
        if (wave == 0 && ct + 1 < NTILES)      // overlap DMA of ct+1 with compute
            tdm_tile_load((unsigned)(uintptr_t)&tiles[cur ^ 1][0],
                          repsB + (size_t)(Cb + 16) * DIM);
#else
        __syncthreads();
        ((v16bf*)tiles[cur])[t] = ((const v16bf*)(repsB + (size_t)Cb * DIM))[t];
        __syncthreads();
#endif
        const unsigned short* tileB = tiles[cur];

        v8f acc = {};
        #pragma unroll
        for (int s = 0; s < 8; ++s) {
            v16bf bfrag;
            ((v8bf*)&bfrag)[0] = *(const v8bf*)(tileB + lr * DIM + 32 * s + 8 * hi);
            ((v8bf*)&bfrag)[1] = *(const v8bf*)(tileB + lr * DIM + 32 * s + 16 + 8 * hi);
            acc = __builtin_amdgcn_wmma_f32_16x16x32_bf16(
                false, afrag[s], false, bfrag, (short)0, acc, false, false);
        }

        if (ct != diagT && ct != posT) {       // scalar-uniform branch
            // hot path: every element is a plain negative (510/512 tiles)
            #pragma unroll
            for (int v = 0; v < 8; ++v) {
                const float x = acc[v];
                const float M = smax(m[v], x);
                ssum[v] = ssum[v] * EXP2F(m[v] - M) + EXP2F(x - M);
                m[v]    = M;
            }
        } else {
            // general path: diagonal exclusion and/or positive capture
            const int j = Cb + lr;
            #pragma unroll
            for (int v = 0; v < 8; ++v) {
                const int   i      = R + v + 8 * hi;
                const float x      = acc[v];
                const bool  isDiag = (j == i);
                const bool  isPos  = (j == ((i + HALF_B) & (NROWS - 1)));
                const float xl     = isDiag ? NEGBIG : x;
                const float M      = smax(m[v], xl);
                ssum[v] = ssum[v] * EXP2F(m[v] - M) + EXP2F(xl - M);
                m[v]    = M;
                pos[v]  = isPos ? x : pos[v];
            }
        }
    }

    // ---- butterfly across the 16 lanes sharing this hi-half ----
    #pragma unroll
    for (int mask = 1; mask < 16; mask <<= 1) {
        #pragma unroll
        for (int v = 0; v < 8; ++v) {
            float mo = __shfl_xor(m[v],    mask, 32);
            float so = __shfl_xor(ssum[v], mask, 32);
            float po = __shfl_xor(pos[v],  mask, 32);
            float M  = smax(m[v], mo);
            ssum[v]  = ssum[v] * EXP2F(m[v] - M) + so * EXP2F(mo - M);
            m[v]     = M;
            pos[v]   = smax(pos[v], po);    // exactly one lane held the real value
        }
    }

    float lossAcc = 0.0f, accAcc = 0.0f;
    if (lr == 0) {
        #pragma unroll
        for (int v = 0; v < 8; ++v) {
            float lse2 = m[v] + LOG2F(ssum[v]);        // log2-domain LSE
            lossAcc += (lse2 - pos[v]) * LN2;          // natural-log loss
            accAcc  += (pos[v] == m[v]) ? 1.0f : 0.0f; // pos is the row max
        }
    }
    lossAcc += __shfl_xor(lossAcc, 16, 32);
    accAcc  += __shfl_xor(accAcc, 16, 32);
    if (lane == 0) { red[wave][0] = lossAcc; red[wave][1] = accAcc; }
    __syncthreads();
    if (t == 0) {
        float L = 0.0f, A = 0.0f;
        #pragma unroll
        for (int w = 0; w < 8; ++w) { L += red[w][0]; A += red[w][1]; }
        partials[blockIdx.x * 2 + 0] = L;
        partials[blockIdx.x * 2 + 1] = A;
    }
}

// ---------------------------------------------------------------------------
// Kernel 3: deterministic final reduction of 64 workgroup partials.
// ---------------------------------------------------------------------------
__global__ void ntxent_finalize(const float* __restrict__ partials,
                                float* __restrict__ out) {
    if (threadIdx.x == 0 && blockIdx.x == 0) {
        float L = 0.0f, A = 0.0f;
        for (int k = 0; k < 64; ++k) { L += partials[2 * k]; A += partials[2 * k + 1]; }
        out[0] = L / (float)NROWS;   // loss / (2b)
        out[1] = A / (float)NROWS;   // accuracy
    }
}

extern "C" void kernel_launch(void* const* d_in, const int* in_sizes, int n_in,
                              void* d_out, int out_size, void* d_ws, size_t ws_size,
                              hipStream_t stream) {
    const float* zis = (const float*)d_in[0];
    const float* zjs = (const float*)d_in[1];
    unsigned short* repsB = (unsigned short*)d_ws;                       // 4 MB bf16
    float* partials = (float*)((char*)d_ws + (size_t)NROWS * DIM * 2);   // 64*2 f32
    float* out = (float*)d_out;

    ntxent_convert<<<(NROWS * DIM + 255) / 256, 256, 0, stream>>>(zis, zjs, repsB);
    ntxent_main<<<NTILES / 8, 256, 0, stream>>>(repsB, partials);
    ntxent_finalize<<<1, 64, 0, stream>>>(partials, out);
}
